// DeformableSelfAttention_14053132992845
// MI455X (gfx1250) — matrix operand order repaired and checked by reading, not compiled, not run on hardware
//
#include <hip/hip_runtime.h>
#include <stdint.h>

// ---------------- problem constants (B==1 forced by reference) ----------------
#define NTOK  16384      // H0*W0
#define DM    512        // d_model (== K for every GEMM in this module)
#define NHD   8          // heads
#define HD    64         // head dim
#define NPTS  7          // sample points
#define GW    128        // grid W
#define GH    128        // grid H

typedef __attribute__((ext_vector_type(8)))  float  v8f;
typedef __attribute__((ext_vector_type(4)))  float  v4f;
typedef __attribute__((ext_vector_type(16))) __bf16 v16bf;
typedef __attribute__((ext_vector_type(8)))  __bf16 v8bf;

// ---------------- bf16 helpers (round-to-nearest-even) ----------------
__device__ __forceinline__ uint16_t f2bf(float f) {
    uint32_t u = __float_as_uint(f);
    u += 0x7FFFu + ((u >> 16) & 1u);
    return (uint16_t)(u >> 16);
}
__device__ __forceinline__ float bf2f(uint16_t h) {
    return __uint_as_float(((uint32_t)h) << 16);
}

// ---------------- elementwise f32 -> bf16 (4 elems/thread) ----------------
__global__ __launch_bounds__(256)
void cvt_bf16_kernel(const float* __restrict__ src, uint16_t* __restrict__ dst, int n4) {
    int i = blockIdx.x * blockDim.x + threadIdx.x;
    if (i >= n4) return;
    v4f v = *reinterpret_cast<const v4f*>(src + 4 * i);
    uint16_t o0 = f2bf(v.x), o1 = f2bf(v.y), o2 = f2bf(v.z), o3 = f2bf(v.w);
    uint64_t packed = (uint64_t)o0 | ((uint64_t)o1 << 16) |
                      ((uint64_t)o2 << 32) | ((uint64_t)o3 << 48);
    *reinterpret_cast<uint64_t*>(dst + 4 * i) = packed;
}

// transpose-convert: src is (rows x cols) f32 row-major; dst is (cols x rows) bf16 row-major
__global__ __launch_bounds__(256)
void trcvt_bf16_kernel(const float* __restrict__ src, uint16_t* __restrict__ dst,
                       int rows, int cols) {
    int t = blockIdx.x * blockDim.x + threadIdx.x;
    if (t >= rows * cols) return;
    int k = t / cols, n = t % cols;
    dst[n * rows + k] = f2bf(src[t]);
}

// ---------------- WMMA bf16 GEMM: C = A(MxK) * B(KxN) + bias, K = DM fixed ----
// A: bf16 row-major (ld = K).  Bt: bf16, B transposed => (N x K) row-major (ld = K).
// MODE: 0 = store fp32, 1 = store bf16, 2 = relu + store bf16
// Block = 128 threads (4 waves).  Block tile 64(M) x 64(N); each wave: 16(M) x 64(N).
template <int MODE>
__global__ __launch_bounds__(128)
void gemm_bf16_wmma_kernel(const uint16_t* __restrict__ A,
                           const uint16_t* __restrict__ Bt,
                           const float* __restrict__ bias,
                           void* __restrict__ C,
                           int M, int Ncols) {
    constexpr int K = DM;
    const int lane = threadIdx.x & 31;
    const int wid  = threadIdx.x >> 5;
    const int m0   = blockIdx.y * 64 + wid * 16;     // 16 rows for this wave
    const int n0   = blockIdx.x * 64;                // 64 cols for this block

    const int lrow  = lane & 15;                     // row / col within 16-group
    const int ahalf = (lane >> 4) ? 16 : 0;          // A: K base per lane half
    const int bhalf = (lane >> 4) ? 16 : 0;          // B: K base per lane half

    v8f acc[4];
#pragma unroll
    for (int t = 0; t < 4; ++t) acc[t] = (v8f){0.f,0.f,0.f,0.f,0.f,0.f,0.f,0.f};

    const uint16_t* arow = A + (size_t)(m0 + lrow) * K + ahalf;
    const uint16_t* bptr[4];
#pragma unroll
    for (int t = 0; t < 4; ++t)
        bptr[t] = Bt + (size_t)(n0 + 16 * t + lrow) * K + bhalf;

#pragma unroll 2
    for (int k0 = 0; k0 < K; k0 += 32) {
        // A fragment: 16x32 bf16.  lane<16: elems0-15 = K k0..k0+15
        //                          lane>=16: elems0-15 = K k0+16..k0+31
        const v8bf* pa = reinterpret_cast<const v8bf*>(arow + k0);
        v8bf alo = pa[0];
        v8bf ahi = pa[1];   // +8 bf16 elements
        v16bf afrag = __builtin_shufflevector(alo, ahi,
                         0,1,2,3,4,5,6,7, 8,9,10,11,12,13,14,15);
        __builtin_prefetch(arow + k0 + 64, 0, 1);    // speculative; dropped if OOB

#pragma unroll
        for (int t = 0; t < 4; ++t) {
            // B fragment: 16 contiguous K values of column n0+16t+lrow
            v16bf bfrag = *reinterpret_cast<const v16bf*>(bptr[t] + k0);
            acc[t] = __builtin_amdgcn_wmma_f32_16x16x32_bf16(
                         false, afrag, false, bfrag, (short)0, acc[t], false, false);
        }
    }

    // Epilogue: C/D layout: vgpr r -> row m0 + r + 8*(lane>=16); col = n0+16t+lrow
    const int mbase = m0 + ((lane >> 4) << 3);
#pragma unroll
    for (int t = 0; t < 4; ++t) {
        const int ncol = n0 + 16 * t + lrow;
        const float b = bias[ncol];
#pragma unroll
        for (int r = 0; r < 8; ++r) {
            float v = acc[t][r] + b;
            const size_t o = (size_t)(mbase + r) * Ncols + ncol;
            if (MODE == 0) {
                ((float*)C)[o] = v;
            } else {
                if (MODE == 2) v = fmaxf(v, 0.f);
                ((uint16_t*)C)[o] = f2bf(v);
            }
        }
    }
}

// ---------------- offset head + sample-index computation ----------------
// off = tanh(H1 @ Woff2 + boff2); samp = coords + off*(2/128); idx = iy*W + ix
// one thread per (token, point)
__global__ __launch_bounds__(256)
void offsets_kernel(const uint16_t* __restrict__ H1b,     // (N x D) bf16, relu'd
                    const uint16_t* __restrict__ W2t,     // (2P x D) bf16 (Woff2^T)
                    const float* __restrict__ boff2,      // (2P)
                    int* __restrict__ idx) {
    int t = blockIdx.x * blockDim.x + threadIdx.x;
    if (t >= NTOK * NPTS) return;
    const int n = t / NPTS, p = t % NPTS;

    const uint16_t* hrow = H1b + (size_t)n * DM;
    const uint16_t* w0   = W2t + (size_t)(2 * p + 0) * DM;
    const uint16_t* w1   = W2t + (size_t)(2 * p + 1) * DM;
    float a0 = boff2[2 * p + 0];
    float a1 = boff2[2 * p + 1];
    for (int k = 0; k < DM; ++k) {
        const float h = bf2f(hrow[k]);
        a0 = fmaf(h, bf2f(w0[k]), a0);
        a1 = fmaf(h, bf2f(w1[k]), a1);
    }
    const float scale = 2.0f / (float)GW;           // max(H,W)=128
    const float offx = tanhf(a0), offy = tanhf(a1);
    const int cx = n & (GW - 1), cy = n >> 7;
    int ix = (int)rintf((float)cx + offx * scale);  // rint = round-half-even (jnp.round)
    int iy = (int)rintf((float)cy + offy * scale);
    ix = min(max(ix, 0), GW - 1);
    iy = min(max(iy, 0), GH - 1);
    idx[t] = iy * GW + ix;
}

// ---------------- deformable attention core ----------------
// one wave32 per (token, head); lane covers 2 of 64 head dims
__global__ __launch_bounds__(256)
void attn_kernel(const uint16_t* __restrict__ Qb,
                 const uint16_t* __restrict__ Kb,
                 const uint16_t* __restrict__ Vb,
                 const int* __restrict__ idx,
                 uint16_t* __restrict__ outh) {
    const int lane = threadIdx.x & 31;
    const int wid  = threadIdx.x >> 5;
    const int g    = blockIdx.x * 8 + wid;          // (token, head) flat id
    const int n    = g >> 3;
    const int h    = g & 7;
    const int d0   = lane * 2;

    const size_t qoff = (size_t)n * DM + h * HD + d0;
    const float q0 = bf2f(Qb[qoff]);
    const float q1 = bf2f(Qb[qoff + 1]);

    float sc[NPTS], v0[NPTS], v1[NPTS];
#pragma unroll
    for (int p = 0; p < NPTS; ++p) {
        const int m = idx[n * NPTS + p];
        const size_t kb = (size_t)m * DM + h * HD + d0;
        const float k0 = bf2f(Kb[kb]);
        const float k1 = bf2f(Kb[kb + 1]);
        v0[p] = bf2f(Vb[kb]);
        v1[p] = bf2f(Vb[kb + 1]);
        float part = q0 * k0 + q1 * k1;
#pragma unroll
        for (int s = 16; s >= 1; s >>= 1) part += __shfl_xor(part, s, 32);
        sc[p] = part * 0.125f;                      // 1/sqrt(64)
    }

    float mx = sc[0];
#pragma unroll
    for (int p = 1; p < NPTS; ++p) mx = fmaxf(mx, sc[p]);
    float sum = 0.f;
#pragma unroll
    for (int p = 0; p < NPTS; ++p) { sc[p] = __expf(sc[p] - mx); sum += sc[p]; }
    const float inv = 1.f / sum;

    float o0 = 0.f, o1 = 0.f;
#pragma unroll
    for (int p = 0; p < NPTS; ++p) { o0 = fmaf(sc[p], v0[p], o0); o1 = fmaf(sc[p], v1[p], o1); }
    outh[qoff]     = f2bf(o0 * inv);
    outh[qoff + 1] = f2bf(o1 * inv);
}

// ---------------- host-side launch ----------------
static inline size_t align256(size_t x) { return (x + 255) & ~(size_t)255; }

extern "C" void kernel_launch(void* const* d_in, const int* in_sizes, int n_in,
                              void* d_out, int out_size, void* d_ws, size_t ws_size,
                              hipStream_t stream) {
    (void)in_sizes; (void)n_in; (void)out_size; (void)ws_size;
    const float* x     = (const float*)d_in[0];
    const float* Wq    = (const float*)d_in[3];
    const float* bq    = (const float*)d_in[4];
    const float* Wk    = (const float*)d_in[5];
    const float* bk    = (const float*)d_in[6];
    const float* Wv    = (const float*)d_in[7];
    const float* bv    = (const float*)d_in[8];
    const float* Wo    = (const float*)d_in[9];
    const float* bo    = (const float*)d_in[10];
    const float* Woff1 = (const float*)d_in[11];
    const float* boff1 = (const float*)d_in[12];
    const float* Woff2 = (const float*)d_in[13];
    const float* boff2 = (const float*)d_in[14];

    // workspace carve-out
    uint8_t* p = (uint8_t*)d_ws;
    size_t cur = 0;
    auto take = [&](size_t bytes) { uint8_t* r = p + cur; cur += align256(bytes); return r; };
    uint16_t* xb   = (uint16_t*)take((size_t)NTOK * DM * 2);
    uint16_t* Wqt  = (uint16_t*)take((size_t)DM * DM * 2);
    uint16_t* Wkt  = (uint16_t*)take((size_t)DM * DM * 2);
    uint16_t* Wvt  = (uint16_t*)take((size_t)DM * DM * 2);
    uint16_t* Wo1t = (uint16_t*)take((size_t)DM * DM * 2);
    uint16_t* Wot  = (uint16_t*)take((size_t)DM * DM * 2);
    uint16_t* W2t  = (uint16_t*)take((size_t)(2 * NPTS) * DM * 2);
    uint16_t* Qb   = (uint16_t*)take((size_t)NTOK * DM * 2);
    uint16_t* Kb   = (uint16_t*)take((size_t)NTOK * DM * 2);
    uint16_t* Vb   = (uint16_t*)take((size_t)NTOK * DM * 2);
    uint16_t* H1b  = (uint16_t*)take((size_t)NTOK * DM * 2);
    int*      idxb = (int*)     take((size_t)NTOK * NPTS * 4);
    uint16_t* outh = (uint16_t*)take((size_t)NTOK * DM * 2);

    // 1) convert activations / transpose-convert weights to bf16
    {
        int n4 = (NTOK * DM) / 4;
        cvt_bf16_kernel<<<(n4 + 255) / 256, 256, 0, stream>>>(x, xb, n4);
        int e = DM * DM;
        trcvt_bf16_kernel<<<(e + 255) / 256, 256, 0, stream>>>(Wq,    Wqt,  DM, DM);
        trcvt_bf16_kernel<<<(e + 255) / 256, 256, 0, stream>>>(Wk,    Wkt,  DM, DM);
        trcvt_bf16_kernel<<<(e + 255) / 256, 256, 0, stream>>>(Wv,    Wvt,  DM, DM);
        trcvt_bf16_kernel<<<(e + 255) / 256, 256, 0, stream>>>(Woff1, Wo1t, DM, DM);
        trcvt_bf16_kernel<<<(e + 255) / 256, 256, 0, stream>>>(Wo,    Wot,  DM, DM);
        int e2 = DM * 2 * NPTS;
        trcvt_bf16_kernel<<<(e2 + 255) / 256, 256, 0, stream>>>(Woff2, W2t, DM, 2 * NPTS);
    }

    // 2) WMMA projections: Q, K, V (bf16 out) and ReLU hidden (bf16 out)
    {
        dim3 grid(DM / 64, NTOK / 64);
        gemm_bf16_wmma_kernel<1><<<grid, 128, 0, stream>>>(xb, Wqt,  bq,    Qb,  NTOK, DM);
        gemm_bf16_wmma_kernel<1><<<grid, 128, 0, stream>>>(xb, Wkt,  bk,    Kb,  NTOK, DM);
        gemm_bf16_wmma_kernel<1><<<grid, 128, 0, stream>>>(xb, Wvt,  bv,    Vb,  NTOK, DM);
        gemm_bf16_wmma_kernel<2><<<grid, 128, 0, stream>>>(xb, Wo1t, boff1, H1b, NTOK, DM);
    }

    // 3) offsets -> gather indices
    {
        int n = NTOK * NPTS;
        offsets_kernel<<<(n + 255) / 256, 256, 0, stream>>>(H1b, W2t, boff2, idxb);
    }

    // 4) deformable attention (wave per token-head)
    attn_kernel<<<(NTOK * NHD) / 8, 256, 0, stream>>>(Qb, Kb, Vb, idxb, outh);

    // 5) output projection -> fp32 d_out
    {
        dim3 grid(DM / 64, NTOK / 64);
        gemm_bf16_wmma_kernel<0><<<grid, 128, 0, stream>>>(outh, Wot, bo, (float*)d_out,
                                                           NTOK, DM);
    }
}
